// ValueFunction_7988639171333
// MI455X (gfx1250) — matrix operand order repaired and checked by reading, not compiled, not run on hardware
//
#include <hip/hip_runtime.h>
#include <hip/hip_bf16.h>

typedef __attribute__((ext_vector_type(16))) __bf16 v16bf;
typedef __attribute__((ext_vector_type(8)))  __bf16 v8bf;
typedef __attribute__((ext_vector_type(8)))  float  v8f;
typedef __attribute__((ext_vector_type(2)))  float  v2f;

#define GAMMA 0.99f
#define TN 48           // pixels per workgroup (half an image row)

// d_out flat offsets (floats), concatenated tuple return order
#define Q_OFF  0
#define V_OFF  2359296
#define R_OFF  2654208
#define VA_OFF 5013504
#define AL_OFF 7372800
#define AT_OFF 9732096
#define TM_OFF 10027008

// workspace layout (float offsets)
#define WS_VBUF0 0
#define WS_VBUF1 294912
#define WS_TMA   589824      // 16x12 f32 padded trans A-matrix
#define WS_PB2   590016      // 2x16 f32 padded stage-2 biases
#define WS_BF16  590048      // bf16 region: pw1[2][128][96], then pw2[2][16][128]

__device__ __forceinline__ int iclamp(int v, int lo, int hi){
  return (v < lo) ? lo : ((v > hi) ? hi : v);
}
__device__ __forceinline__ float softplus_f(float x){ return (x > 20.f) ? x : log1pf(expf(x)); }
__device__ __forceinline__ float mish_f(float x){ return x * tanhf(softplus_f(x)); }
__device__ __forceinline__ float sigmoid_f(float x){ return 1.f / (1.f + expf(-x)); }

__device__ __forceinline__ v8f wmma_bf16(v16bf a, v16bf b, v8f c){
  return __builtin_amdgcn_wmma_f32_16x16x32_bf16(false, a, false, b, (short)0, c, false, false);
}
// A-matrix lane fragment: elements 0..15 = base[0..15]
__device__ __forceinline__ v16bf loadA_pw(const __bf16* base){
  v8bf lo = *(const v8bf*)base;
  v8bf hi = *(const v8bf*)(base + 8);
  return __builtin_shufflevector(lo, hi, 0,1,2,3,4,5,6,7,8,9,10,11,12,13,14,15);
}

// ---------------- Kernel 0: softmax of motion kernels + padded A ------------
__global__ void k_trans(const float* __restrict__ w, float* __restrict__ out_tm,
                        float* __restrict__ tmA){
  int a = threadIdx.x;
  if (a < 8){
    float v[9]; float mx = -1e30f;
    #pragma unroll
    for(int i=0;i<9;i++){ v[i] = w[a*9+i]*10.0f; mx = fmaxf(mx, v[i]); }
    float s = 0.f;
    #pragma unroll
    for(int i=0;i<9;i++){ v[i] = expf(v[i]-mx); s += v[i]; }
    float inv = 1.f/s;
    #pragma unroll
    for(int i=0;i<9;i++){ float r = v[i]*inv; out_tm[a*9+i] = r; tmA[a*12+i] = r; }
    tmA[a*12+9] = 0.f; tmA[a*12+10] = 0.f; tmA[a*12+11] = 0.f;
  } else if (a < 16){
    #pragma unroll
    for(int i=0;i<12;i++) tmA[a*12+i] = 0.f;
  }
}

// ---------------- Kernel 0b: pad + convert weights to bf16 ------------------
__global__ __launch_bounds__(256)
void k_prep(const float* __restrict__ vw1, const float* __restrict__ rw1,
            const float* __restrict__ vw2, const float* __restrict__ rw2,
            const float* __restrict__ vb2, const float* __restrict__ rb2,
            __bf16* __restrict__ pw1, __bf16* __restrict__ pw2,
            float* __restrict__ pb2){
  int idx = blockIdx.x*256 + threadIdx.x;
  if (idx < 2*128*96){                        // pw1[net][m][k], k padded 72->96
    int net = idx / (128*96);
    int rem = idx % (128*96);
    int m = rem / 96, k = rem % 96;
    const float* W1 = net ? rw1 : vw1;
    int kc = (k < 72) ? k : 0;
    float raw = W1[m*72 + kc];
    pw1[idx] = (__bf16)((k < 72) ? raw : 0.f);
  } else {
    int i2 = idx - 2*128*96;
    if (i2 < 2*16*128){                       // pw2[net][r][k], rows padded ->16
      int net = i2 / 2048;
      int rem = i2 % 2048;
      int r = rem / 128, k = rem % 128;
      const float* W2 = net ? rw2 : vw2;
      int rows = net ? 8 : 9;
      int rc = (r < rows) ? r : 0;
      float raw = W2[rc*128 + k];
      pw2[i2] = (__bf16)((r < rows) ? raw : 0.f);
    } else {
      int i3 = i2 - 2*16*128;
      if (i3 < 32){                           // pb2[net][m]
        int net = i3 / 16, m = i3 % 16;
        const float* B2 = net ? rb2 : vb2;
        int rows = net ? 8 : 9;
        int mc = (m < rows) ? m : 0;
        float raw = B2[mc];
        pb2[i3] = (m < rows) ? raw : 0.f;
      }
    }
  }
}

// ---------------- Kernel 1: fused conv front-end ----------------------------
__global__ __launch_bounds__(256)
void k_frontend(const float* __restrict__ env,
                const __bf16* __restrict__ pw1, const __bf16* __restrict__ pw2,
                const float* __restrict__ vb1, const float* __restrict__ rb1,
                const float* __restrict__ pb2,
                const float* __restrict__ rfail_p,
                float* __restrict__ out, float* __restrict__ v0buf){
  __shared__ __attribute__((aligned(32))) __bf16 Blds[TN*96];      // [n][k]
  __shared__ __attribute__((aligned(32))) __bf16 hLds[2][TN*128];  // [net][n][m]
  __shared__ float oLds[2][16*TN];                                 // [net][row][n]
  __shared__ float rLds[8*TN];                                     // reward cache

  const int bid = blockIdx.x;            // 0..6143
  const int b   = bid / 192;
  const int rem = bid % 192;
  const int y   = rem >> 1;
  const int x0  = (rem & 1) * TN;

  const int tid  = threadIdx.x;
  const int wave = tid >> 5;
  const int lane = tid & 31;
  const int lr   = lane & 15;
  const int kofs = (lane < 16) ? 0 : 16;  // A-matrix K sub-offset
  const int bofs = (lane < 16) ? 0 : 16;  // B-matrix K sub-offset

  // ---- im2col B in LDS (clamp + select, no divergent loads) ----
  for (int idx = tid; idx < TN*72; idx += 256){
    int n = idx / 72, k = idx % 72;
    int c = k / 9, r = k % 9, ky = r / 3, kx = r % 3;
    int yy = y + ky - 1, xx = x0 + n + kx - 1;
    int yc = iclamp(yy, 0, 95), xc = iclamp(xx, 0, 95);
    float raw = env[((b*8 + c)*96 + yc)*96 + xc];
    float val = (yy == yc && xx == xc) ? raw : 0.f;
    Blds[n*96 + k] = (__bf16)val;
  }
  for (int idx = tid; idx < TN*24; idx += 256){
    int n = idx / 24, k = 72 + idx % 24;
    Blds[n*96 + k] = (__bf16)0.f;
  }
  __syncthreads();

  // ---- stage 1: each wave owns a 16-channel M-tile for both nets ----
  const int row = wave*16 + lr;   // hidden channel
  v8f accV[3], accR[3];
  #pragma unroll
  for (int net = 0; net < 2; net++){
    const __bf16* ar = pw1 + (net*128 + row)*96;
    v16bf a0 = loadA_pw(ar +  0 + kofs);
    v16bf a1 = loadA_pw(ar + 32 + kofs);
    v16bf a2 = loadA_pw(ar + 64 + kofs);
    #pragma unroll
    for (int nt = 0; nt < 3; nt++){
      int n = nt*16 + lr;
      v8f c = {};
      c = wmma_bf16(a0, *(const v16bf*)&Blds[n*96 +  0 + bofs], c);
      c = wmma_bf16(a1, *(const v16bf*)&Blds[n*96 + 32 + bofs], c);
      c = wmma_bf16(a2, *(const v16bf*)&Blds[n*96 + 64 + bofs], c);
      if (net) accR[nt] = c; else accV[nt] = c;
    }
  }

  // bias + mish + pack to LDS as [n][m] (B-matrix of stage 2)
  #pragma unroll
  for (int net = 0; net < 2; net++){
    const float* bias = net ? rb1 : vb1;
    #pragma unroll
    for (int nt = 0; nt < 3; nt++){
      v8f c = net ? accR[nt] : accV[nt];
      int n  = nt*16 + lr;
      int m0 = wave*16 + ((lane < 16) ? 0 : 8);
      v8bf hv;
      #pragma unroll
      for (int r = 0; r < 8; r++)
        hv[r] = (__bf16)mish_f(c[r] + bias[m0 + r]);
      *(v8bf*)&hLds[net][n*128 + m0] = hv;   // 16B contiguous store
    }
  }
  __syncthreads();

  // ---- stage 2: 1x1 conv, 6 wave-tasks (net x n-tile) ----
  if (wave < 6){
    int net = wave / 3, nt = wave % 3;
    const __bf16* ar = pw2 + (net*16 + lr)*128;
    v8f c = {};
    #pragma unroll
    for (int s = 0; s < 4; s++){           // K = 128
      v16bf a  = loadA_pw(ar + 32*s + kofs);
      int n = nt*16 + lr;
      v16bf bv = *(const v16bf*)&hLds[net][n*128 + 32*s + bofs];
      c = wmma_bf16(a, bv, c);
    }
    int n  = nt*16 + lr;
    int r0 = (lane < 16) ? 0 : 8;
    #pragma unroll
    for (int r = 0; r < 8; r++){
      int m = r0 + r;
      oLds[net][m*TN + n] = c[r] + pb2[net*16 + m];
    }
  }
  __syncthreads();

  // ---- epilogue ----
  const float rfail = -softplus_f(rfail_p[0]);
  for (int idx = tid; idx < 8*TN; idx += 256){
    int a = idx / TN, n = idx % TN;
    float logit = oLds[0][a*TN + n];
    float th    = oLds[0][8*TN + n];
    float va    = sigmoid_f(logit - th);
    float rm    = oLds[1][a*TN + n];
    float rew   = rfail*(1.f - va) + rm*va;
    rLds[a*TN + n] = rew;
    int g = b*73728 + a*9216 + y*96 + x0 + n;
    out[AL_OFF + g] = logit;
    out[VA_OFF + g] = va;
    out[R_OFF  + g] = rew;
  }
  for (int n = tid; n < TN; n += 256){
    int g1 = b*9216 + y*96 + x0 + n;
    out[AT_OFF + g1] = oLds[0][8*TN + n];
  }
  __syncthreads();
  for (int n = tid; n < TN; n += 256){
    float vmax = -1e30f;
    #pragma unroll
    for (int a = 0; a < 8; a++) vmax = fmaxf(vmax, rLds[a*TN + n]);
    v0buf[b*9216 + y*96 + x0 + n] = vmax;   // v0 = max(reward)
  }
}

// ---------------- Kernel 2: one value-iteration step ------------------------
__global__ __launch_bounds__(256)
void k_vi(const float* __restrict__ vin, float* __restrict__ vnext,
          const float* __restrict__ tmA, float* __restrict__ out, int last){
  const int tid  = threadIdx.x;
  const int wave = tid >> 5;
  const int lane = tid & 31;
  const int lr   = lane & 15;
  const int hi   = (lane >= 16);

  const int t   = blockIdx.x*8 + wave;    // tile id, 0..18431
  const int b   = t / 576;
  const int rem = t % 576;
  const int y   = rem / 6;
  const int x0  = (rem % 6) * 16;

  const float* vb = vin + b*9216;
  v8f c = {};
  #pragma unroll
  for (int s = 0; s < 3; s++){
    int k0 = 4*s + (hi ? 2 : 0);
    v2f Av = *(const v2f*)&tmA[lr*12 + k0];   // padded, unconditional b64 load
    v2f Bv;
    #pragma unroll
    for (int j = 0; j < 2; j++){
      int k  = k0 + j;
      int kk = (k < 9) ? k : 0;
      int ky = kk / 3, kx = kk % 3;
      int yy = y + ky - 1, xx = x0 + lr + kx - 1;
      int yc = iclamp(yy, 0, 95), xc = iclamp(xx, 0, 95);
      float raw = vb[yc*96 + xc];
      Bv[j] = (k < 9 && yy == yc && xx == xc) ? raw : 0.f;
    }
    c = __builtin_amdgcn_wmma_f32_16x16x4_f32(false, Av, false, Bv,
                                              (short)0, c, false, false);
  }

  if (lane < 16){
    int x = x0 + lr;
    int g1 = b*9216 + y*96 + x;
    float q[8]; float vmax = -1e30f;
    #pragma unroll
    for (int a = 0; a < 8; a++){
      int g = b*73728 + a*9216 + y*96 + x;
      float rew = out[R_OFF + g];
      float va  = out[VA_OFF + g];
      q[a] = rew + GAMMA * c[a] * va;
      vmax = fmaxf(vmax, q[a]);
    }
    if (last){
      #pragma unroll
      for (int a = 0; a < 8; a++)
        out[Q_OFF + b*73728 + a*9216 + y*96 + x] = q[a];
      out[V_OFF + g1] = vmax;
    } else {
      vnext[g1] = vmax;
    }
  }
}

extern "C" void kernel_launch(void* const* d_in, const int* in_sizes, int n_in,
                              void* d_out, int out_size, void* d_ws, size_t ws_size,
                              hipStream_t stream) {
  const float* env  = (const float*)d_in[0];
  const float* wtm  = (const float*)d_in[1];
  const float* rfp  = (const float*)d_in[2];
  const float* vw1  = (const float*)d_in[3];
  const float* vb1  = (const float*)d_in[4];
  const float* vw2  = (const float*)d_in[5];
  const float* vb2  = (const float*)d_in[6];
  const float* rw1  = (const float*)d_in[7];
  const float* rb1  = (const float*)d_in[8];
  const float* rw2  = (const float*)d_in[9];
  const float* rb2  = (const float*)d_in[10];
  float* out = (float*)d_out;
  float* ws  = (float*)d_ws;

  float*  vbuf0 = ws + WS_VBUF0;
  float*  vbuf1 = ws + WS_VBUF1;
  float*  tmA   = ws + WS_TMA;
  float*  pb2   = ws + WS_PB2;
  __bf16* pw1   = (__bf16*)(ws + WS_BF16);
  __bf16* pw2   = pw1 + 2*128*96;

  k_trans<<<1, 32, 0, stream>>>(wtm, out + TM_OFF, tmA);
  k_prep<<<112, 256, 0, stream>>>(vw1, rw1, vw2, rw2, vb2, rb2, pw1, pw2, pb2);
  k_frontend<<<6144, 256, 0, stream>>>(env, pw1, pw2, vb1, rb1, pb2, rfp, out, vbuf0);
  for (int i = 0; i < 40; i++){
    const float* vi_in  = (i & 1) ? vbuf1 : vbuf0;
    float*       vi_out = (i & 1) ? vbuf0 : vbuf1;
    k_vi<<<2304, 256, 0, stream>>>(vi_in, vi_out, tmA, out, (i == 39) ? 1 : 0);
  }
}
